// SpectralFreqTimeConv1D_7172595384340
// MI455X (gfx1250) — matrix operand order repaired and checked by reading, not compiled, not run on hardware
//
#include <hip/hip_runtime.h>
#include <hip/hip_bf16.h>

// ---------------------------------------------------------------------------
// SpectralFreqTimeConv1D on MI455X (gfx1250, wave32, WMMA)
//   fwd truncated DFT (f16 WMMA, f32 acc)  ->  per-mode complex matmul +
//   time modulation (f32 WMMA)             ->  truncated inverse DFT (f16 WMMA)
// Software-pipelined K-loops, branch-free speculative prefetch.
// ---------------------------------------------------------------------------

typedef _Float16 v16h __attribute__((ext_vector_type(16)));
typedef _Float16 v8h  __attribute__((ext_vector_type(8)));
typedef _Float16 v2h  __attribute__((ext_vector_type(2)));
typedef float    v8f  __attribute__((ext_vector_type(8)));
typedef float    v2f  __attribute__((ext_vector_type(2)));

union U16 { v16h v; v8h h[2]; };

#define B_    32
#define GRID_ 4096
#define CI_   64
#define CO_   64
#define TE_   256
#define MH_   257      // retained rFFT modes
#define MPAD_ 272      // 17 * 16
#define KPAD_ 544      // 2*257=514 padded to 17*32

__device__ __forceinline__ float theta_of(int m, int g) {
    // exact phase: 2*pi * ((m*g) mod 4096) / 4096
    int p = (m * g) & (GRID_ - 1);
    return (float)p * (6.283185307179586f / (float)GRID_);
}

// ---- table builders (run every launch; deterministic) ----------------------
__global__ void build_fwd_tables(_Float16* __restrict__ fcos,
                                 _Float16* __restrict__ fsin) {
    int idx = blockIdx.x * blockDim.x + threadIdx.x;
    if (idx >= MPAD_ * GRID_) return;
    int m = idx >> 12;            // / 4096
    int g = idx & (GRID_ - 1);
    float c = 0.f, s = 0.f;
    if (m < MH_) {
        float th = theta_of(m, g);
        c =  cosf(th) * (1.0f / (float)GRID_);   // norm="forward": 1/N
        s = -sinf(th) * (1.0f / (float)GRID_);   // Xi = sum (-sin/N) * x
    }
    fcos[idx] = (_Float16)c;
    fsin[idx] = (_Float16)s;
}

__global__ void build_inv_table(_Float16* __restrict__ ainv) {
    int idx = blockIdx.x * blockDim.x + threadIdx.x;
    if (idx >= GRID_ * KPAD_) return;
    int g = idx / KPAD_;
    int k = idx - g * KPAD_;
    float v = 0.f;
    if (k < 2 * MH_) {
        int m = k >> 1;
        float th = theta_of(m, g);
        float cm = (m == 0) ? 1.0f : 2.0f;       // hermitian doubling
        v = (k & 1) ? (-cm * sinf(th)) : (cm * cosf(th));
    }
    ainv[idx] = (_Float16)v;
}

__global__ void zero_zpad(_Float16* __restrict__ z) {
    int idx = blockIdx.x * blockDim.x + threadIdx.x;
    const int nrows = B_ * CO_;
    const int npad  = KPAD_ - 2 * MH_;           // 30
    if (idx >= nrows * npad) return;
    int row = idx / npad;
    int k   = 2 * MH_ + (idx - row * npad);
    z[row * KPAD_ + k] = (_Float16)0.f;
}

// ---- x (b,g,i) f32  ->  xT (b,i,g) f16  via LDS tile -----------------------
__global__ void transpose_x(const float* __restrict__ x, _Float16* __restrict__ xT) {
    __shared__ float tile[32][33];
    int b  = blockIdx.z;
    int g0 = blockIdx.x * 32;
    int i0 = blockIdx.y * 32;
    int tx = threadIdx.x, ty = threadIdx.y;
    tile[ty][tx] = x[((size_t)b * GRID_ + (g0 + ty)) * CI_ + (i0 + tx)];
    __syncthreads();
    xT[((size_t)b * CI_ + (i0 + ty)) * GRID_ + (g0 + tx)] = (_Float16)tile[tx][ty];
}

// ---- time modulation: t_tr = t_emb @ (dense_real, dense_imag) --------------
__global__ void tmod(const float* __restrict__ temb, const float* __restrict__ dr,
                     const float* __restrict__ di, float* __restrict__ trp,
                     float* __restrict__ tip) {
    int m    = blockIdx.x;         // 0..256
    int b    = threadIdx.x & 31;
    int part = threadIdx.x >> 5;   // 0 = real, 1 = imag
    const float* d = part ? di : dr;
    float s = 0.f;
    for (int k = 0; k < TE_; ++k) s += temb[b * TE_ + k] * d[k * MH_ + m];
    (part ? tip : trp)[b * MH_ + m] = s;
}

// ---- operand tile loaders (ISA 7.12.2 per-lane layouts, pure b128 loads) ---
__device__ __forceinline__ void load_fwd_tile(const _Float16* __restrict__ ac,
                                              const _Float16* __restrict__ as,
                                              const _Float16* __restrict__ bx,
                                              int kb, int off1,
                                              U16& a0, U16& a1, U16& bb) {
    a0.h[0] = *(const v8h*)(ac + kb + off1);
    a0.h[1] = *(const v8h*)(ac + kb + off1 + 16);
    a1.h[0] = *(const v8h*)(as + kb + off1);
    a1.h[1] = *(const v8h*)(as + kb + off1 + 16);
    bb.h[0] = *(const v8h*)(bx + kb);
    bb.h[1] = *(const v8h*)(bx + kb + 16);
}

__device__ __forceinline__ void load_inv_tile(const _Float16* __restrict__ ar,
                                              const _Float16* __restrict__ zb,
                                              int kb, int off1,
                                              U16& aa, U16& bb) {
    aa.h[0] = *(const v8h*)(ar + kb + off1);
    aa.h[1] = *(const v8h*)(ar + kb + off1 + 16);
    bb.h[0] = *(const v8h*)(zb + kb);
    bb.h[1] = *(const v8h*)(zb + kb + 16);
}

// ---- forward truncated DFT: XF[m][b][i] = F[m][g] * xT[b][i][g] ------------
// per block: one (b, m-tile); 4 waves = 4 i-tiles. K = 4096, f16 WMMA K=32.
// Two-stage software pipeline: next tile's loads issued before current WMMAs.
__global__ void fwd_dft(const _Float16* __restrict__ fcos,
                        const _Float16* __restrict__ fsin,
                        const _Float16* __restrict__ xT,
                        float* __restrict__ xfr, float* __restrict__ xfi) {
    const int mtile = blockIdx.x;            // 0..16
    const int b     = blockIdx.y;            // 0..31
    const int wave  = threadIdx.x >> 5;      // i-tile 0..3
    const int lane  = threadIdx.x & 31;
    const int lh    = lane & 15;
    const int hi    = lane >> 4;
    const int m     = mtile * 16 + lh;       // A row (per ISA layout)
    const int i     = wave * 16 + lh;        // B col
    const int off1  = hi * 8;                // A K-offset per lane half

    const _Float16* ac = fcos + (size_t)m * GRID_;
    const _Float16* as = fsin + (size_t)m * GRID_;
    const _Float16* bx = xT + ((size_t)b * CI_ + i) * GRID_ + hi * 8;

    v8f accR = {}; v8f accI = {};
    U16 a0A, a1A, bbA, a0B, a1B, bbB;
    load_fwd_tile(ac, as, bx, 0, off1, a0A, a1A, bbA);
    for (int kb = 0; kb < GRID_; kb += 64) {
        // stage B loads for kb+32 before consuming stage A (kb)
        load_fwd_tile(ac, as, bx, kb + 32, off1, a0B, a1B, bbB);
        __builtin_prefetch(ac + kb + 512, 0, 1);   // unconditional, speculative
        __builtin_prefetch(as + kb + 512, 0, 1);
        __builtin_prefetch(bx + kb + 512, 0, 1);
        accR = __builtin_amdgcn_wmma_f32_16x16x32_f16(false, a0A.v, false, bbA.v,
                                                      (short)0, accR, false, false);
        accI = __builtin_amdgcn_wmma_f32_16x16x32_f16(false, a1A.v, false, bbA.v,
                                                      (short)0, accI, false, false);
        if (kb + 64 < GRID_)
            load_fwd_tile(ac, as, bx, kb + 64, off1, a0A, a1A, bbA);
        accR = __builtin_amdgcn_wmma_f32_16x16x32_f16(false, a0B.v, false, bbB.v,
                                                      (short)0, accR, false, false);
        accI = __builtin_amdgcn_wmma_f32_16x16x32_f16(false, a1B.v, false, bbB.v,
                                                      (short)0, accI, false, false);
    }
#pragma unroll
    for (int r = 0; r < 8; ++r) {
        int mr = mtile * 16 + r + hi * 8;    // D row mapping (VGPR r)
        size_t o = ((size_t)mr * B_ + b) * CI_ + i;
        xfr[o] = accR[r];
        xfi[o] = accI[r];
    }
}

// ---- per-mode complex matmul (f32 WMMA 16x16x4) + time modulation ----------
// One block per mode; 8 waves = {btile 0..1} x {otile 0..3}.  M=32,N=64,K=64.
__global__ void mode_mm(const float* __restrict__ xfr, const float* __restrict__ xfi,
                        const float* __restrict__ wr_, const float* __restrict__ wi_,
                        const float* __restrict__ trp, const float* __restrict__ tip,
                        _Float16* __restrict__ z) {
    const int m     = blockIdx.x;            // 0..256
    const int wave  = threadIdx.x >> 5;
    const int btile = wave >> 2;             // 0..1
    const int otile = wave & 3;              // 0..3
    const int lane  = threadIdx.x & 31;
    const int lh    = lane & 15;
    const int hi    = lane >> 4;
    const int brow  = btile * 16 + lh;       // A row = batch
    const int o     = otile * 16 + lh;       // B col = out channel
    const int koff  = hi * 2;                // f32 WMMA K split per lane half

    v8f accR = {}; v8f accI = {};
#pragma unroll 4
    for (int k0 = 0; k0 < CI_; k0 += 4) {
        int ka = k0 + koff;
        v2f ar = *(const v2f*)(xfr + ((size_t)m * B_ + brow) * CI_ + ka);
        v2f ai = *(const v2f*)(xfi + ((size_t)m * B_ + brow) * CI_ + ka);
        const float* pr = wr_ + ((size_t)m * CI_ + ka) * CO_ + o;
        const float* pi = wi_ + ((size_t)m * CI_ + ka) * CO_ + o;
        v2f br; br.x = pr[0]; br.y = pr[CO_];
        v2f bi; bi.x = pi[0]; bi.y = pi[CO_];
        v2f nbi; nbi.x = -bi.x; nbi.y = -bi.y;   // f32 WMMA has no A/B NEG
        accR = __builtin_amdgcn_wmma_f32_16x16x4_f32(false, ar, false, br,
                                                     (short)0, accR, false, false);
        accR = __builtin_amdgcn_wmma_f32_16x16x4_f32(false, ai, false, nbi,
                                                     (short)0, accR, false, false);
        accI = __builtin_amdgcn_wmma_f32_16x16x4_f32(false, ar, false, bi,
                                                     (short)0, accI, false, false);
        accI = __builtin_amdgcn_wmma_f32_16x16x4_f32(false, ai, false, br,
                                                     (short)0, accI, false, false);
    }
#pragma unroll
    for (int r = 0; r < 8; ++r) {
        int bb = btile * 16 + r + hi * 8;
        float tr = trp[bb * MH_ + m];
        float ti = tip[bb * MH_ + m];
        float yr = accR[r] * tr - accI[r] * ti;
        float yi = accR[r] * ti + accI[r] * tr;
        v2h pk; pk.x = (_Float16)yr; pk.y = (_Float16)yi;
        *(v2h*)(z + ((size_t)bb * CO_ + o) * KPAD_ + 2 * m) = pk;
    }
}

// ---- inverse truncated DFT: out[b][g][o] = Ainv[g][k] * Z[b][o][k] ---------
// 17 K-tiles, fully unrolled with ping-pong operand registers.
__global__ void inv_dft(const _Float16* __restrict__ ainv,
                        const _Float16* __restrict__ z,
                        float* __restrict__ out) {
    const int gtile = blockIdx.x;            // 0..255
    const int b     = blockIdx.y;            // 0..31
    const int wave  = threadIdx.x >> 5;      // o-tile 0..3
    const int lane  = threadIdx.x & 31;
    const int lh    = lane & 15;
    const int hi    = lane >> 4;
    const int g     = gtile * 16 + lh;
    const int o     = wave * 16 + lh;
    const int off1  = hi * 8;

    const _Float16* ar = ainv + (size_t)g * KPAD_;
    const _Float16* zb = z + ((size_t)b * CO_ + o) * KPAD_ + hi * 8;

    v8f acc = {};
    U16 aa[2], bb[2];
    load_inv_tile(ar, zb, 0, off1, aa[0], bb[0]);
#pragma unroll
    for (int t = 0; t < KPAD_ / 32; ++t) {   // 17 iterations, unrolled
        const int cur = t & 1;
        if (t + 1 < KPAD_ / 32)
            load_inv_tile(ar, zb, (t + 1) * 32, off1, aa[cur ^ 1], bb[cur ^ 1]);
        acc = __builtin_amdgcn_wmma_f32_16x16x32_f16(false, aa[cur].v, false, bb[cur].v,
                                                     (short)0, acc, false, false);
    }
#pragma unroll
    for (int r = 0; r < 8; ++r) {
        int gr = gtile * 16 + r + hi * 8;
        out[((size_t)b * GRID_ + gr) * CO_ + o] = acc[r];
    }
}

// ---------------------------------------------------------------------------
extern "C" void kernel_launch(void* const* d_in, const int* in_sizes, int n_in,
                              void* d_out, int out_size, void* d_ws, size_t ws_size,
                              hipStream_t stream) {
    const float* x    = (const float*)d_in[0];   // (32,4096,64)
    const float* temb = (const float*)d_in[1];   // (32,256)
    const float* wr   = (const float*)d_in[2];   // (257,64,64)
    const float* wi   = (const float*)d_in[3];   // (257,64,64)
    const float* dr   = (const float*)d_in[4];   // (256,257)
    const float* di   = (const float*)d_in[5];   // (256,257)
    float* out = (float*)d_out;

    char* ws = (char*)d_ws;
    size_t off = 0;
    _Float16* fcos = (_Float16*)(ws + off); off += (size_t)MPAD_ * GRID_ * 2;
    _Float16* fsin = (_Float16*)(ws + off); off += (size_t)MPAD_ * GRID_ * 2;
    _Float16* ainv = (_Float16*)(ws + off); off += (size_t)GRID_ * KPAD_ * 2;
    _Float16* xT   = (_Float16*)(ws + off); off += (size_t)B_ * CI_ * GRID_ * 2;
    float*    xfr  = (float*)   (ws + off); off += (size_t)MPAD_ * B_ * CI_ * 4;
    float*    xfi  = (float*)   (ws + off); off += (size_t)MPAD_ * B_ * CI_ * 4;
    _Float16* zbuf = (_Float16*)(ws + off); off += (size_t)B_ * CO_ * KPAD_ * 2;
    float*    trp  = (float*)   (ws + off); off += (size_t)B_ * MH_ * 4;
    float*    tip  = (float*)   (ws + off); off += (size_t)B_ * MH_ * 4;
    // total ~32.5 MB of d_ws

    build_fwd_tables<<<(MPAD_ * GRID_ + 255) / 256, 256, 0, stream>>>(fcos, fsin);
    build_inv_table<<<(GRID_ * KPAD_ + 255) / 256, 256, 0, stream>>>(ainv);
    zero_zpad<<<(B_ * CO_ * (KPAD_ - 2 * MH_) + 255) / 256, 256, 0, stream>>>(zbuf);
    transpose_x<<<dim3(GRID_ / 32, CI_ / 32, B_), dim3(32, 32), 0, stream>>>(x, xT);
    tmod<<<MH_, 64, 0, stream>>>(temb, dr, di, trp, tip);
    fwd_dft<<<dim3(MPAD_ / 16, B_), 128, 0, stream>>>(fcos, fsin, xT, xfr, xfi);
    mode_mm<<<MH_, 256, 0, stream>>>(xfr, xfi, wr, wi, trp, tip, zbuf);
    inv_dft<<<dim3(GRID_ / 16, B_), 128, 0, stream>>>(ainv, zbuf, out);
}
